// LstmSeqMnist_3813930959150
// MI455X (gfx1250) — matrix operand order repaired and checked by reading, not compiled, not run on hardware
//
#include <hip/hip_runtime.h>

// LSTM (B=128, T=512, I=64, H=512) + linear head (C=10) for MI455X / gfx1250.
//  - prep kernel: W_ih, W_hh fp32 -> bf16 in workspace; bias = b_ih + b_hh.
//  - persistent kernel: 8 workgroups x 8 wave32; each WG owns 16 batch rows
//    for all 512 steps (LSTM recurrence is batch-independent).
//    Each wave owns the SAME 64-wide hidden slice in all four gates, so the
//    i/f/g/o accumulators align element-wise and the cell update is
//    register-only. h tile (16x512 bf16) is the WMMA A operand in LDS.
//  - x_t prefetch for t+1 is issued AFTER the pre-compute barrier (barriers
//    drain loadcnt, so issuing before would stall); it completes during the
//    WMMA phase.
//  - gate nonlinearities use the CDNA5 hardware V_TANH_F32 transcendental
//    (sigmoid(x) = 0.5 + 0.5*tanh(x/2)).

#define Bb   128
#define Tt   512
#define Ii   64
#define Hh   512
#define G4   (4 * Hh)   // 2048
#define Cc   10

typedef __attribute__((ext_vector_type(16))) __bf16 v16bf;
typedef __attribute__((ext_vector_type(8)))  float  v8f;

union Frag {
    uint4 u[2];
    v16bf v;
};

__device__ __forceinline__ float fast_tanh(float x) {
#if __has_builtin(__builtin_amdgcn_tanhf)
    return __builtin_amdgcn_tanhf(x);
#else
    float r;
    // v_nop covers the TRANS32 result-use hazard the scheduler can't see.
    asm volatile("v_tanh_f32 %0, %1\n\tv_nop" : "=v"(r) : "v"(x));
    return r;
#endif
}

__device__ __forceinline__ float fast_sigmoid(float x) {
    return __builtin_fmaf(0.5f, fast_tanh(0.5f * x), 0.5f);
}

// ---------------------------------------------------------------------------
// Prep: fp32 weights -> bf16, fuse biases.
// ---------------------------------------------------------------------------
__global__ __launch_bounds__(256)
void lstm_prep(const float* __restrict__ Wih_f, const float* __restrict__ Whh_f,
               const float* __restrict__ bih,   const float* __restrict__ bhh,
               __bf16* __restrict__ Wih_b, __bf16* __restrict__ Whh_b,
               float* __restrict__ bias) {
    const int idx = blockIdx.x * blockDim.x + threadIdx.x;
    if (idx < G4 * Ii) Wih_b[idx] = (__bf16)Wih_f[idx];
    if (idx < G4 * Hh) Whh_b[idx] = (__bf16)Whh_f[idx];
    if (idx < G4)      bias[idx]  = bih[idx] + bhh[idx];
}

// ---------------------------------------------------------------------------
// Persistent LSTM kernel: grid = 8 blocks x 256 threads (8 wave32).
// ---------------------------------------------------------------------------
__global__ __launch_bounds__(256, 1)
void lstm_persist(const float* __restrict__ x,
                  const __bf16* __restrict__ Wih,
                  const __bf16* __restrict__ Whh,
                  const float* __restrict__ bias,
                  const float* __restrict__ Wcls,
                  const float* __restrict__ bcls,
                  float* __restrict__ out) {
    __shared__ __bf16 sH [16 * Hh];   // 16 KB: hidden state tile (A operand)
    __shared__ __bf16 sX [16 * Ii];   // 2 KB : current x_t tile (A operand)
    __shared__ float  sHf[16 * Hh];   // 32 KB: fp32 h_last for the classifier

    const int tid   = threadIdx.x;
    const int lane  = tid & 31;
    const int wv    = tid >> 5;       // wave id 0..7 -> hidden slice [wv*64, wv*64+64)
    const int b0    = blockIdx.x * 16;
    const int nlo   = lane & 15;      // n (B/C col) or m (A row)
    const int khalf = lane >> 4;      // which K-half / M-half a lane covers

    // x staging coordinates for this thread (16 rows x 64 cols, 4 floats/thread)
    const int xrow = tid >> 4;
    const int xcol = (tid & 15) * 4;
    const float* xbase = &x[((size_t)(b0 + xrow)) * ((size_t)Tt * Ii) + xcol];

    // zero h
    for (int i = tid; i < 16 * Hh / 2; i += 256)
        ((unsigned int*)sH)[i] = 0u;

    // per-lane gate bias: n = g*512 + wv*64 + nt*16 + nlo (constant over rows m)
    float biasv[4][4];
#pragma unroll
    for (int g = 0; g < 4; ++g)
#pragma unroll
        for (int nt = 0; nt < 4; ++nt)
            biasv[g][nt] = bias[g * Hh + wv * 64 + nt * 16 + nlo];

    // cell state in registers: creg[nt][r] <-> (m = r + 8*khalf, j = wv*64+nt*16+nlo)
    float creg[4][8];
#pragma unroll
    for (int nt = 0; nt < 4; ++nt)
#pragma unroll
        for (int r = 0; r < 8; ++r) creg[nt][r] = 0.0f;

    // prefetch x for t = 0
    float4 xreg = *(const float4*)&xbase[0];

    __syncthreads();

    for (int t = 0; t < Tt; ++t) {
        // ---- stage x_t tile (registers -> LDS bf16)
        {
            __bf16* dst = &sX[xrow * Ii + xcol];
            dst[0] = (__bf16)xreg.x; dst[1] = (__bf16)xreg.y;
            dst[2] = (__bf16)xreg.z; dst[3] = (__bf16)xreg.w;
        }
        __syncthreads();

        // ---- issue x prefetch for t+1 AFTER the barrier; it completes
        //      during the WMMA phase and is drained by the cheap post-compute
        //      barrier instead of stalling here.
        {
            const int tn = (t + 1 < Tt) ? (t + 1) : t;   // clamped redundant load
            xreg = *(const float4*)&xbase[(size_t)tn * Ii];
        }

        const v8f zero = {0.f, 0.f, 0.f, 0.f, 0.f, 0.f, 0.f, 0.f};
        v8f acc[4][4];
#pragma unroll
        for (int g = 0; g < 4; ++g)
#pragma unroll
            for (int nt = 0; nt < 4; ++nt) acc[g][nt] = zero;

        // ---- input projection: A = sX (16x64 bf16), B = Wih rows (ld = 64)
        //      (compiler hoists the 32 W_ih B-fragments across the t loop)
#pragma unroll
        for (int kt = 0; kt < Ii / 32; ++kt) {
            const int k0 = kt * 32;
            Frag a;
            a.u[0] = *(const uint4*)&sX[nlo * Ii + k0 + khalf * 8];
            a.u[1] = *(const uint4*)&sX[nlo * Ii + k0 + 16 + khalf * 8];
#pragma unroll
            for (int g = 0; g < 4; ++g) {
#pragma unroll
                for (int nt = 0; nt < 4; ++nt) {
                    const int nrow = g * Hh + wv * 64 + nt * 16 + nlo;
                    const __bf16* bp = &Wih[(size_t)nrow * Ii + k0 + khalf * 8];
                    Frag b;
                    b.u[0] = *(const uint4*)bp;
                    b.u[1] = *(const uint4*)(bp + 16);
                    acc[g][nt] = __builtin_amdgcn_wmma_f32_16x16x32_bf16(
                        false, a.v, false, b.v, (short)0, acc[g][nt], false, false);
                }
            }
        }

        // ---- recurrent GEMM: A = sH (16x512 bf16), B = Whh rows (ld = 512)
#pragma unroll 4
        for (int kt = 0; kt < Hh / 32; ++kt) {
            const int k0 = kt * 32;
            Frag a;
            a.u[0] = *(const uint4*)&sH[nlo * Hh + k0 + khalf * 8];
            a.u[1] = *(const uint4*)&sH[nlo * Hh + k0 + 16 + khalf * 8];
#pragma unroll
            for (int g = 0; g < 4; ++g) {
#pragma unroll
                for (int nt = 0; nt < 4; ++nt) {
                    const int nrow = g * Hh + wv * 64 + nt * 16 + nlo;
                    const __bf16* bp = &Whh[(size_t)nrow * Hh + k0 + khalf * 8];
                    Frag b;
                    b.u[0] = *(const uint4*)bp;
                    b.u[1] = *(const uint4*)(bp + 16);
                    acc[g][nt] = __builtin_amdgcn_wmma_f32_16x16x32_bf16(
                        false, a.v, false, b.v, (short)0, acc[g][nt], false, false);
                }
            }
        }

        __syncthreads();   // everyone done *reading* sH before we overwrite it

        // ---- gates -> cell update -> new h (register-only; h written bf16)
        const bool last = (t == Tt - 1);
#pragma unroll
        for (int nt = 0; nt < 4; ++nt) {
            const int j = wv * 64 + nt * 16 + nlo;
#pragma unroll
            for (int r = 0; r < 8; ++r) {
                const int m = r + 8 * khalf;
                const float iv = fast_sigmoid(acc[0][nt][r] + biasv[0][nt]);
                const float fv = fast_sigmoid(acc[1][nt][r] + biasv[1][nt]);
                const float gv = fast_tanh   (acc[2][nt][r] + biasv[2][nt]);
                const float ov = fast_sigmoid(acc[3][nt][r] + biasv[3][nt]);
                const float cv = __builtin_fmaf(fv, creg[nt][r], iv * gv);
                creg[nt][r] = cv;
                const float hv = ov * fast_tanh(cv);
                sH[m * Hh + j] = (__bf16)hv;
                if (last) sHf[m * Hh + j] = hv;   // fp32 h_last for the head
            }
        }
        __syncthreads();   // h visible before next step's WMMA reads
    }

    // ---- classifier head: 16 rows x 10 classes, one fp32 dot per thread
    if (tid < 16 * Cc) {
        const int m   = tid / Cc;
        const int cls = tid % Cc;
        float sum = bcls[cls];
        const float* hrow = &sHf[m * Hh];
        const float* wrow = &Wcls[(size_t)cls * Hh];
        for (int k = 0; k < Hh; ++k)
            sum += hrow[k] * wrow[k];
        out[(b0 + m) * Cc + cls] = sum;
    }
}

// ---------------------------------------------------------------------------
extern "C" void kernel_launch(void* const* d_in, const int* in_sizes, int n_in,
                              void* d_out, int out_size, void* d_ws, size_t ws_size,
                              hipStream_t stream) {
    const float* x    = (const float*)d_in[0];
    const float* Wih  = (const float*)d_in[1];
    const float* Whh  = (const float*)d_in[2];
    const float* bih  = (const float*)d_in[3];
    const float* bhh  = (const float*)d_in[4];
    const float* Wcls = (const float*)d_in[5];
    const float* bcls = (const float*)d_in[6];
    float* out = (float*)d_out;

    char* ws = (char*)d_ws;
    __bf16* Whh_b = (__bf16*)ws;                                  // 2 MB
    __bf16* Wih_b = (__bf16*)(ws + (size_t)G4 * Hh * 2);          // 256 KB
    float*  bias  = (float*)(ws + (size_t)G4 * Hh * 2 + (size_t)G4 * Ii * 2);

    lstm_prep<<<(G4 * Hh + 255) / 256, 256, 0, stream>>>(
        Wih, Whh, bih, bhh, Wih_b, Whh_b, bias);

    lstm_persist<<<Bb / 16, 256, 0, stream>>>(
        x, Wih_b, Whh_b, bias, Wcls, bcls, out);
}
